// TopKSAE_85529978732862
// MI455X (gfx1250) — matrix operand order repaired and checked by reading, not compiled, not run on hardware
//
#include <hip/hip_runtime.h>
#include <hip/hip_bf16.h>

// ---------------------------------------------------------------------------
// TopK SAE for MI455X (gfx1250, wave32, WMMA)
//   encode: z = x @ We^T   (bf16 WMMA, f32 accum, LDS double-buffered,
//                           tiles staged by the Tensor Data Mover (TDM))
//   topk  : per-row top-32 by |z|, mask z in place, record (idx,val)
//   decode: xhat = sparse_z @ Wd^T  via pre-transposed Wd (coalesced gathers)
// ---------------------------------------------------------------------------

#define D_MODEL 1024
#define M_LAT   16384
#define N_TOK   8192
#define TOPK    32

// encode tiling
#define TOK_BLK 64                 // tokens per block
#define LAT_BLK 256                // latents per block
#define KSTEP   32                 // bf16 WMMA K depth
#define NKS     (D_MODEL / KSTEP)  // 32 K-steps
// LDS: rows padded 32 -> 40 bf16 (80B). TDM pad_enable reproduces this layout:
// pad_interval=3 (64B between pads), pad_amount=3 (4 DWORDs = 16B).
#define ROW_PAD   40
#define A_ELEMS   (TOK_BLK * ROW_PAD)              // 2560 bf16
#define B_OFF     A_ELEMS
#define BUF_ELEMS (A_ELEMS + LAT_BLK * ROW_PAD)    // 12800 bf16 (25.6KB/buffer)

typedef __attribute__((ext_vector_type(16))) __bf16 v16bf;
typedef __attribute__((ext_vector_type(8)))  __bf16 v8bf;
typedef __attribute__((ext_vector_type(8)))  float  v8f;
typedef __attribute__((ext_vector_type(4)))  unsigned int u32x4;
typedef __attribute__((ext_vector_type(8)))  int          i32x8;
typedef __attribute__((ext_vector_type(4)))  int          i32x4;

union Frag16 { v16bf v; v8bf h[2]; };

// Generic pointers into __shared__ carry the LDS byte offset in their low 32
// bits (shared aperture base has zero low bits).
__device__ __forceinline__ unsigned lds_off(const void* p) {
    return (unsigned)(size_t)p;
}

// ---- Tensor Data Mover: 2D tile load, global -> padded LDS rows ------------
// D# per CDNA5 ISA ch.8: group0 = {count,lds_addr,global_addr,type=2},
// group1 = {data_size=2B, pad_enable, pad_interval, pad_amount, dims, strides}.
// This toolchain exposes the 6-arg builtin:
//   (u32x4 g0, i32x8 g1, i32x4 g2, i32x4 g3, i32x8 extra, i32 cpol)
__device__ __forceinline__ void tdm_load_tile_2d(unsigned lds_addr,
                                                 const void* gaddr,
                                                 int rows /* tile_dim1 */) {
    unsigned long long ga = (unsigned long long)gaddr;
    u32x4 g0;
    g0.x = 1u;                                     // count=1 (valid), user mode
    g0.y = lds_addr;                               // [63:32] lds_addr
    g0.z = (unsigned)(ga & 0xFFFFFFFFu);           // global_addr[31:0]
    g0.w = (unsigned)((ga >> 32) & 0x01FFFFFFu)    // global_addr[56:32]
         | (2u << 30);                             // [127:126] type=2 (image)
    i32x8 g1;
    g1[0] = (1 << 16)                              // data_size=1 -> 2 bytes
          | (1 << 20)                              // pad_enable
          | (3 << 22)                              // pad_interval: 16 DW = 64B
          | (3 << 25);                             // pad_amount:   4 DW = 16B
    g1[1] = (KSTEP & 0xFFFF) << 16;                // tensor_dim0[15:0] = 32
    g1[2] = (rows & 0xFFFF) << 16;                 // td0 hi=0 | tensor_dim1 lo
    g1[3] = (KSTEP & 0xFFFF) << 16;                // td1 hi=0 | tile_dim0 = 32
    g1[4] = rows & 0xFFFF;                         // tile_dim1 | tile_dim2=0
    g1[5] = D_MODEL;                               // tensor_dim0_stride lo
    g1[6] = 0;                                     // stride hi | dim1_stride lo
    g1[7] = 0;
    i32x4 zero4 = {0, 0, 0, 0};
    i32x8 zero8 = {0, 0, 0, 0, 0, 0, 0, 0};
    __builtin_amdgcn_tensor_load_to_lds(g0, g1, zero4, zero4, zero8, 0);
}

// ---------------------------------------------------------------- f32 -> bf16
__device__ __forceinline__ unsigned short f32_to_bf16_rne(float f) {
    unsigned int u = __float_as_uint(f);
    u += 0x7FFFu + ((u >> 16) & 1u);   // round-to-nearest-even
    return (unsigned short)(u >> 16);
}

__global__ __launch_bounds__(256)
void cvt_bf16_kernel(const float* __restrict__ in, unsigned short* __restrict__ out,
                     size_t count) {
    size_t i = (size_t)blockIdx.x * 256 + threadIdx.x;
    if (i < count) out[i] = f32_to_bf16_rne(in[i]);
}

// ---------------------------------------------------------------- encode GEMM
// z[N_TOK, M_LAT] = x[N_TOK, D] @ We[M_LAT, D]^T   (NT: both K-contiguous)
// Block: 256 thr = 8 waves = 2 (token) x 4 (latent); block tile 64 tok x 256 lat.
// Wave tile: 32 tok x 64 lat = 2 A frags x 4 B frags = 8 WMMAs per K-step.
// Wave 0 drives the TDM: 2 descriptors per K-step stage the 20KB tile pair
// into the inactive LDS buffer while all 8 waves run WMMAs on the active one.
__global__ __launch_bounds__(256)
void encode_wmma_kernel(const unsigned short* __restrict__ xb,
                        const unsigned short* __restrict__ web,
                        float* __restrict__ z) {
    __shared__ unsigned short lds[2][BUF_ELEMS];   // 51.2 KB of 320 KB WGP LDS

    const int t    = threadIdx.x;
    const int lane = t & 31;
    const int wave = t >> 5;
    const int ln   = lane & 15;
    const int hi   = lane >> 4;          // 0: lanes 0-15, 1: lanes 16-31
    const int wTok = wave & 1;           // token sub-block (2 x 32)
    const int wLat = wave >> 1;          // latent sub-block (4 x 64)

    const int tok0 = blockIdx.y * TOK_BLK;
    const int lat0 = blockIdx.x * LAT_BLK;

    // uniform tile bases for the TDM descriptors
    const unsigned short* aTile = xb  + (size_t)tok0 * D_MODEL;   // 64 rows
    const unsigned short* bTile = web + (size_t)lat0 * D_MODEL;   // 256 rows
    unsigned aDst[2] = { lds_off(&lds[0][0]),     lds_off(&lds[1][0]) };
    unsigned bDst[2] = { lds_off(&lds[0][B_OFF]), lds_off(&lds[1][B_OFF]) };

    // ---- LDS fragment pointers for this wave ----
    // A frag (ISA 16x32 layout): lanes0-15 K{0..7,16..23}, lanes16-31 +8.
    const unsigned short* lA0base = &lds[0][(wTok * 32 + ln) * ROW_PAD + hi * 8];
    // B frag: lanes0-15 K{0..7,16..23}, lanes16-31 +8, column = lat row.
    const unsigned short* lB0base = &lds[0][B_OFF + (wLat * 64 + ln) * ROW_PAD + hi * 8];

    v8f acc[2][4];
    #pragma unroll
    for (int i = 0; i < 2; ++i)
        #pragma unroll
        for (int j = 0; j < 4; ++j) acc[i][j] = (v8f){};

    // ---- prologue: TDM-stage K-step 0 into buffer 0 (wave 0 only) ----
    if (wave == 0) {
        tdm_load_tile_2d(aDst[0], aTile, TOK_BLK);
        tdm_load_tile_2d(bDst[0], bTile, LAT_BLK);
        __builtin_amdgcn_s_wait_tensorcnt((short)0);
    }
    __syncthreads();

    int buf = 0;
    for (int ks = 0; ks < NKS; ++ks) {
        // -- wave 0 launches DMA for the next K-step into the other buffer --
        const bool more = (ks + 1) < NKS;
        if (more && wave == 0) {
            const int k0n = (ks + 1) * KSTEP;
            const int nb  = buf ^ 1;
            tdm_load_tile_2d(aDst[nb], aTile + k0n, TOK_BLK);
            tdm_load_tile_2d(bDst[nb], bTile + k0n, LAT_BLK);
        }

        // -- compute 8 WMMAs from LDS buffer `buf` (TDM drains meanwhile) --
        const unsigned short* lA = lA0base + buf * BUF_ELEMS;
        const unsigned short* lB = lB0base + buf * BUF_ELEMS;

        Frag16 a0, a1;
        a0.h[0] = *(const v8bf*)(lA);
        a0.h[1] = *(const v8bf*)(lA + 16);
        a1.h[0] = *(const v8bf*)(lA + 16 * ROW_PAD);
        a1.h[1] = *(const v8bf*)(lA + 16 * ROW_PAD + 16);

        #pragma unroll
        for (int bt = 0; bt < 4; ++bt) {
            Frag16 b;
            const unsigned short* p = lB + bt * 16 * ROW_PAD;
            b.h[0] = *(const v8bf*)(p);
            b.h[1] = *(const v8bf*)(p + 16);
            acc[0][bt] = __builtin_amdgcn_wmma_f32_16x16x32_bf16(
                false, a0.v, false, b.v, (short)0, acc[0][bt], false, false);
            acc[1][bt] = __builtin_amdgcn_wmma_f32_16x16x32_bf16(
                false, a1.v, false, b.v, (short)0, acc[1][bt], false, false);
        }

        // wave 0 waits for its DMA, then block-wide visibility
        if (more && wave == 0)
            __builtin_amdgcn_s_wait_tensorcnt((short)0);
        __syncthreads();
        buf ^= 1;
    }

    // ---- epilogue: C/D layout (VGPR r -> row r + hi*8, col ln) ----
    #pragma unroll
    for (int i = 0; i < 2; ++i) {
        float* zrow = z + (size_t)(tok0 + wTok * 32 + i * 16 + hi * 8) * M_LAT
                        + lat0 + wLat * 64 + ln;
        #pragma unroll
        for (int r = 0; r < 8; ++r) {
            zrow[(size_t)r * M_LAT +  0] = acc[i][0][r];
            zrow[(size_t)r * M_LAT + 16] = acc[i][1][r];
            zrow[(size_t)r * M_LAT + 32] = acc[i][2][r];
            zrow[(size_t)r * M_LAT + 48] = acc[i][3][r];
        }
    }
}

// ---------------------------------------------------------------- top-k mask
// One block per token row. Stage |z| in LDS (64KB), 32 deterministic argmax
// passes (mark selected slot with -1), record (idx,val), then mask in place.
__global__ __launch_bounds__(256)
void topk_kernel(float* __restrict__ z, int* __restrict__ idx_out,
                 float* __restrict__ val_out) {
    extern __shared__ float smem[];
    float* absv = smem;                        // [M_LAT]
    float* redV = smem + M_LAT;                // [256]
    int*   redI = (int*)(smem + M_LAT + 256);  // [256]

    const int n = blockIdx.x;
    const size_t base = (size_t)n * M_LAT;
    const int t = threadIdx.x;

    for (int i = t; i < M_LAT; i += 256) absv[i] = fabsf(z[base + i]);
    __syncthreads();

    for (int k = 0; k < TOPK; ++k) {
        float bm = -1.0e30f; int bi = 0;
        for (int i = t; i < M_LAT; i += 256) {
            float v = absv[i];
            if (v > bm) { bm = v; bi = i; }    // strict > : lowest index wins ties
        }
        redV[t] = bm; redI[t] = bi;
        __syncthreads();
        for (int s = 128; s > 0; s >>= 1) {
            if (t < s && redV[t + s] > redV[t]) {
                redV[t] = redV[t + s]; redI[t] = redI[t + s];
            }
            __syncthreads();
        }
        if (t == 0) {
            int sel = redI[0];
            idx_out[n * TOPK + k] = sel;
            val_out[n * TOPK + k] = z[base + sel];
            absv[sel] = -1.0f;                 // marker: selected (|z| >= 0 always)
        }
        __syncthreads();
    }

    for (int i = t; i < M_LAT; i += 256)
        if (absv[i] != -1.0f) z[base + i] = 0.0f;
}

// ---------------------------------------------------------------- Wd -> Wd^T
__global__ __launch_bounds__(256)
void transpose_kernel(const float* __restrict__ src,   // [D_MODEL, M_LAT]
                      float* __restrict__ dst) {       // [M_LAT, D_MODEL]
    __shared__ float tile[32][33];
    const int m0 = blockIdx.x * 32;
    const int d0 = blockIdx.y * 32;
    #pragma unroll
    for (int j = 0; j < 32; j += 8)
        tile[threadIdx.y + j][threadIdx.x] =
            src[(size_t)(d0 + threadIdx.y + j) * M_LAT + m0 + threadIdx.x];
    __syncthreads();
    #pragma unroll
    for (int j = 0; j < 32; j += 8)
        dst[(size_t)(m0 + threadIdx.y + j) * D_MODEL + d0 + threadIdx.x] =
            tile[threadIdx.x][threadIdx.y + j];
}

// ---------------------------------------------------------------- sparse decode
// xhat[n,:] = sum_k val[n,k] * WdT[idx[n,k], :]   (coalesced row gathers)
__global__ __launch_bounds__(256)
void decode_kernel(const float* __restrict__ wdt, const int* __restrict__ idx,
                   const float* __restrict__ val, float* __restrict__ xhat) {
    __shared__ int   sidx[TOPK];
    __shared__ float sval[TOPK];
    const int n = blockIdx.x;
    if (threadIdx.x < TOPK) {
        sidx[threadIdx.x] = idx[n * TOPK + threadIdx.x];
        sval[threadIdx.x] = val[n * TOPK + threadIdx.x];
    }
    __syncthreads();

    float acc[4] = {0.f, 0.f, 0.f, 0.f};
    for (int k = 0; k < TOPK; ++k) {
        const float* row = wdt + (size_t)sidx[k] * D_MODEL;
        const float  v   = sval[k];
        #pragma unroll
        for (int j = 0; j < 4; ++j)
            acc[j] += v * row[threadIdx.x + 256 * j];
    }
    #pragma unroll
    for (int j = 0; j < 4; ++j)
        xhat[(size_t)n * D_MODEL + threadIdx.x + 256 * j] = acc[j];
}

// ---------------------------------------------------------------- launcher
extern "C" void kernel_launch(void* const* d_in, const int* in_sizes, int n_in,
                              void* d_out, int out_size, void* d_ws, size_t ws_size,
                              hipStream_t stream) {
    const float* x  = (const float*)d_in[0];   // [N_TOK, D_MODEL]
    const float* We = (const float*)d_in[1];   // [M_LAT, D_MODEL]
    const float* Wd = (const float*)d_in[2];   // [D_MODEL, M_LAT]

    const size_t ND = (size_t)N_TOK * D_MODEL;   // 8,388,608
    const size_t MD = (size_t)M_LAT * D_MODEL;   // 16,777,216

    float* xhat = (float*)d_out;                 // [N_TOK, D_MODEL]
    float* z    = xhat + ND;                     // [N_TOK, M_LAT] (masked)

    // scratch layout: x_bf16 | We_bf16 | WdT(f32) | idx | val  (~119.5 MB)
    char* ws = (char*)d_ws;
    unsigned short* xb   = (unsigned short*)ws;
    unsigned short* web  = (unsigned short*)(ws + ND * 2);
    float*          wdt  = (float*)(ws + ND * 2 + MD * 2);
    int*            sidx = (int*)  (ws + ND * 2 + MD * 2 + MD * 4);
    float*          sval = (float*)(ws + ND * 2 + MD * 2 + MD * 4 +
                                    (size_t)N_TOK * TOPK * 4);

    // 1) downconvert x, We to bf16
    cvt_bf16_kernel<<<(unsigned)((ND + 255) / 256), 256, 0, stream>>>(x, xb, ND);
    cvt_bf16_kernel<<<(unsigned)((MD + 255) / 256), 256, 0, stream>>>(We, web, MD);

    // 2) encode GEMM: z = x @ We^T
    encode_wmma_kernel<<<dim3(M_LAT / LAT_BLK, N_TOK / TOK_BLK), 256, 0, stream>>>(
        xb, web, z);

    // 3) per-row top-32 mask (dynamic LDS: |z| row + reduction scratch)
    const size_t lds_bytes = (size_t)M_LAT * 4 + 256 * 4 + 256 * 4;  // 67584
    topk_kernel<<<N_TOK, 256, lds_bytes, stream>>>(z, sidx, sval);

    // 4) Wd -> Wd^T for coalesced sparse decode
    transpose_kernel<<<dim3(M_LAT / 32, D_MODEL / 32), dim3(32, 8), 0, stream>>>(Wd, wdt);

    // 5) sparse decode: xhat = z_sparse @ Wd^T
    decode_kernel<<<N_TOK, 256, 0, stream>>>(wdt, sidx, sval, xhat);
}
